// RVQCodec_14989435863673
// MI455X (gfx1250) — compile-verified, matrix operand order checked by reading, not run on hardware
//
#include <hip/hip_runtime.h>
#include <cstdint>

#define TPB 256

typedef __bf16 bf16_t;
typedef bf16_t v16bf __attribute__((ext_vector_type(16)));
typedef bf16_t v8bf  __attribute__((ext_vector_type(8)));
typedef float  v8f   __attribute__((ext_vector_type(8)));

__device__ __forceinline__ float silu_f(float x) { return x / (1.f + __expf(-x)); }

__device__ __forceinline__ unsigned pack_bf16x2(float a, float b) {
  union { unsigned u; bf16_t h[2]; } c;
  c.h[0] = (bf16_t)a; c.h[1] = (bf16_t)b;
  return c.u;
}

// ======================= WMMA GEMM (bf16 in, f32 acc) =======================
// C[M,N] = act(A[M,K] @ W + bias) (+ residual)
// wtrans==0 : W is (K,N) row-major.   wtrans==1 : W is (N,K) row-major (A @ W^T).
// Block: 256 threads = 8 waves (2x4). Block tile 64(M) x 128(N); wave tile 32x32
// = 2x2 of 16x16 -> 4 x v_wmma_f32_16x16x32_bf16 per wave per 32-deep K step.
// LDS tiles are stored k-contiguous per row (A: [m][k], B: transposed [n][k]) with
// 40-element (80B, 16B-aligned) row stride so fragment halves load as ds_load_b128.
#define BM 64
#define BN 128
#define BK 32
#define LSTR 40

__global__ __launch_bounds__(TPB)
void gemm_wmma_kernel(const float* __restrict__ A, const float* __restrict__ W,
                      const float* __restrict__ bias, const float* __restrict__ residual,
                      float* __restrict__ C, int M, int N, int K, int act, int wtrans)
{
  __shared__ bf16_t As[BM * LSTR];
  __shared__ bf16_t Bt[BN * LSTR];

  const int tid  = threadIdx.x;
  const int wave = tid >> 5;
  const int lane = tid & 31;
  const int wm   = wave >> 2;   // 0..1
  const int wn   = wave & 3;    // 0..3
  const int bm   = blockIdx.y * BM;
  const int bn   = blockIdx.x * BN;

  v8f acc00 = {0.f,0.f,0.f,0.f,0.f,0.f,0.f,0.f};
  v8f acc01 = acc00, acc10 = acc00, acc11 = acc00;

  for (int k0 = 0; k0 < K; k0 += BK) {
    // ---- stage A tile (BM x BK): packed bf16 pairs, one b32 LDS store per cell
    for (int t = tid; t < BM * 16; t += TPB) {
      int m = t >> 4, kp = t & 15;               // kp fastest: 8B/lane coalesced
      int gm = bm + m;
      float a0 = 0.f, a1 = 0.f;
      if (gm < M) {
        const float* p = A + (size_t)gm * K + k0 + 2 * kp;
        a0 = p[0]; a1 = p[1];
        if (k0 + BK < K) __builtin_prefetch(p + BK, 0, 0);   // global_prefetch_b8
      }
      *(unsigned*)&As[m * LSTR + 2 * kp] = pack_bf16x2(a0, a1);
    }
    // ---- stage B tile transposed (Bt[n][k]), BN x BK
    if (wtrans) {
      for (int t = tid; t < BN * 16; t += TPB) {
        int n = t >> 4, kp = t & 15;             // kp fastest: contiguous 8B/lane
        const float* p = W + (size_t)(bn + n) * K + k0 + 2 * kp;
        if (k0 + BK < K) __builtin_prefetch(p + BK, 0, 0);
        *(unsigned*)&Bt[n * LSTR + 2 * kp] = pack_bf16x2(p[0], p[1]);
      }
    } else {
      for (int t = tid; t < 16 * BN; t += TPB) {
        int kp = t >> 7, n = t & 127;            // n fastest: coalesced along N
        const float* p = W + (size_t)(k0 + 2 * kp) * N + bn + n;
        if (k0 + BK < K) __builtin_prefetch(p + (size_t)BK * N, 0, 0);
        *(unsigned*)&Bt[n * LSTR + 2 * kp] = pack_bf16x2(p[0], p[N]);
      }
    }
    __syncthreads();

    // ---- fragments (ISA 7.12.2 16-bit 16x32 layout). Per-lane halves are
    //      contiguous in LDS -> two ds_load_b128 per fragment.
    const int kb = ((lane < 16) ? 0 : 8);
    const int r0 = wm * 32 + (lane & 15);
    const int c0 = wn * 32 + (lane & 15);
    v8bf aL0 = *(const v8bf*)&As[r0        * LSTR + kb];
    v8bf aH0 = *(const v8bf*)&As[r0        * LSTR + kb + 16];
    v8bf aL1 = *(const v8bf*)&As[(r0 + 16) * LSTR + kb];
    v8bf aH1 = *(const v8bf*)&As[(r0 + 16) * LSTR + kb + 16];
    v8bf bL0 = *(const v8bf*)&Bt[c0        * LSTR + kb];
    v8bf bH0 = *(const v8bf*)&Bt[c0        * LSTR + kb + 16];
    v8bf bL1 = *(const v8bf*)&Bt[(c0 + 16) * LSTR + kb];
    v8bf bH1 = *(const v8bf*)&Bt[(c0 + 16) * LSTR + kb + 16];
    v16bf a0 = __builtin_shufflevector(aL0, aH0, 0,1,2,3,4,5,6,7,8,9,10,11,12,13,14,15);
    v16bf a1 = __builtin_shufflevector(aL1, aH1, 0,1,2,3,4,5,6,7,8,9,10,11,12,13,14,15);
    v16bf b0 = __builtin_shufflevector(bL0, bH0, 0,1,2,3,4,5,6,7,8,9,10,11,12,13,14,15);
    v16bf b1 = __builtin_shufflevector(bL1, bH1, 0,1,2,3,4,5,6,7,8,9,10,11,12,13,14,15);

    acc00 = __builtin_amdgcn_wmma_f32_16x16x32_bf16(false, a0, false, b0, (short)0, acc00, false, false);
    acc01 = __builtin_amdgcn_wmma_f32_16x16x32_bf16(false, a0, false, b1, (short)0, acc01, false, false);
    acc10 = __builtin_amdgcn_wmma_f32_16x16x32_bf16(false, a1, false, b0, (short)0, acc10, false, false);
    acc11 = __builtin_amdgcn_wmma_f32_16x16x32_bf16(false, a1, false, b1, (short)0, acc11, false, false);
    __syncthreads();
  }

  // ---- epilogue. C/D layout: VGPR j -> M = (lane<16?0:8)+j, N = lane&15.
  const int lm = (lane < 16) ? 0 : 8;
  const int ln = lane & 15;
#pragma unroll
  for (int mi = 0; mi < 2; ++mi) {
#pragma unroll
    for (int ni = 0; ni < 2; ++ni) {
      v8f acc = (mi == 0) ? (ni == 0 ? acc00 : acc01) : (ni == 0 ? acc10 : acc11);
      int n  = bn + wn * 32 + ni * 16 + ln;
      int mb = bm + wm * 32 + mi * 16 + lm;
      float bv = bias ? bias[n] : 0.f;
#pragma unroll
      for (int j = 0; j < 8; ++j) {
        int m = mb + j;
        if (m < M) {
          float v = acc[j] + bv;
          if (act == 1)      v = silu_f(v);
          else if (act == 2) v = tanhf(v);
          if (residual) v += residual[(size_t)m * N + n];
          C[(size_t)m * N + n] = v;
        }
      }
    }
  }
}

// ======================= conv1d / conv_transpose1d =======================
__global__ __launch_bounds__(TPB)
void conv1d_kernel(const float* __restrict__ x, const float* __restrict__ w,
                   const float* __restrict__ bias, const float* __restrict__ residual,
                   float* __restrict__ y, int B, int Cin, int Lin, int Cout, int Lout,
                   int K, int stride, int pad, int dil, int act)
{
  long idx = (long)blockIdx.x * TPB + threadIdx.x;
  long total = (long)B * Cout * Lout;
  if (idx >= total) return;
  int t = (int)(idx % Lout);
  long tmp = idx / Lout;
  int co = (int)(tmp % Cout);
  int b  = (int)(tmp / Cout);
  float sum = bias ? bias[co] : 0.f;
  const float* xb = x + (size_t)b * Cin * Lin;
  const float* wc = w + (size_t)co * Cin * K;
  int t0 = t * stride - pad;
  for (int ci = 0; ci < Cin; ++ci) {
    const float* xc = xb + (size_t)ci * Lin;
    const float* wk = wc + ci * K;
    for (int kk = 0; kk < K; ++kk) {
      int tin = t0 + kk * dil;
      if (tin >= 0 && tin < Lin) sum += xc[tin] * wk[kk];
    }
  }
  if (act == 1)      sum = silu_f(sum);
  else if (act == 2) sum = tanhf(sum);
  if (residual) sum += residual[idx];
  y[idx] = sum;
}

__global__ __launch_bounds__(TPB)
void convt1d_kernel(const float* __restrict__ x, const float* __restrict__ w,
                    const float* __restrict__ bias, float* __restrict__ y,
                    int B, int Cin, int Lin, int Cout, int Lout, int K, int stride, int pad)
{
  long idx = (long)blockIdx.x * TPB + threadIdx.x;
  long total = (long)B * Cout * Lout;
  if (idx >= total) return;
  int t = (int)(idx % Lout);
  long tmp = idx / Lout;
  int co = (int)(tmp % Cout);
  int b  = (int)(tmp / Cout);
  float sum = bias ? bias[co] : 0.f;
  const float* xb = x + (size_t)b * Cin * Lin;
  const float* wc = w + (size_t)co * Cin * K;   // weight (Cout, Cin, K)
  int base = t - (K - 1 - pad);
  for (int ci = 0; ci < Cin; ++ci) {
    const float* xc = xb + (size_t)ci * Lin;
    const float* wk = wc + ci * K;
    for (int kk = 0; kk < K; ++kk) {
      int u = base + kk;                        // coordinate in stride-dilated input
      if (u >= 0 && (u % stride) == 0) {
        int tin = u / stride;
        if (tin < Lin) sum += xc[tin] * wk[kk];
      }
    }
  }
  y[idx] = sum;
}

// ======================= layernorm =======================
__global__ __launch_bounds__(TPB)
void layernorm_kernel(const float* __restrict__ x, const float* __restrict__ w,
                      const float* __restrict__ b, float* __restrict__ y, int M, int D)
{
  int row = blockIdx.x;
  if (row >= M) return;
  __shared__ float red[TPB];
  const float* xr = x + (size_t)row * D;
  float s = 0.f;
  for (int i = threadIdx.x; i < D; i += TPB) s += xr[i];
  red[threadIdx.x] = s; __syncthreads();
  for (int o = TPB / 2; o > 0; o >>= 1) {
    if (threadIdx.x < o) red[threadIdx.x] += red[threadIdx.x + o];
    __syncthreads();
  }
  float mean = red[0] / D; __syncthreads();
  float v = 0.f;
  for (int i = threadIdx.x; i < D; i += TPB) { float d = xr[i] - mean; v += d * d; }
  red[threadIdx.x] = v; __syncthreads();
  for (int o = TPB / 2; o > 0; o >>= 1) {
    if (threadIdx.x < o) red[threadIdx.x] += red[threadIdx.x + o];
    __syncthreads();
  }
  float rstd = rsqrtf(red[0] / D + 1e-5f);
  for (int i = threadIdx.x; i < D; i += TPB)
    y[(size_t)row * D + i] = (xr[i] - mean) * rstd * w[i] + b[i];
}

// ======================= attention (wave32 flash, inline RoPE) =======================
// One wave per (b,h,q). Lane j owns the RoPE pair (2j,2j+1) of hd=64.
// Reference quirk: cos[:H] -> rope angle = h * freq_j (position == head index).
__global__ __launch_bounds__(TPB)
void attn_kernel(const float* __restrict__ q, const float* __restrict__ k,
                 const float* __restrict__ v, float* __restrict__ o,
                 int B, int H, int T, int hd)
{
  int wave = threadIdx.x >> 5, lane = threadIdx.x & 31;
  long wg = (long)blockIdx.x * 8 + wave;
  if (wg >= (long)B * H * T) return;
  int qt = (int)(wg % T);
  long bh = wg / T;
  int h = (int)(bh % H);
  int b = (int)(bh / H);
  int D = H * hd;
  int j = lane;
  float freq = __powf(10000.f, -(float)(2 * j) / (float)hd);
  float sv, cv;
  __sincosf((float)h * freq, &sv, &cv);

  const float* qp = q + ((size_t)(b * T + qt)) * D + h * hd;
  float qe = qp[2 * j], qo = qp[2 * j + 1];
  float qre = qe * cv - qo * sv, qro = qe * sv + qo * cv;
  float scale = rsqrtf((float)hd);

  float m = -3.0e38f, l = 0.f, ae = 0.f, ao = 0.f;
  for (int kt = 0; kt <= qt; ++kt) {           // causal mask
    const float* kp = k + ((size_t)(b * T + kt)) * D + h * hd;
    float ke = kp[2 * j], ko = kp[2 * j + 1];
    float kre = ke * cv - ko * sv, kro = ke * sv + ko * cv;
    float part = qre * kre + qro * kro;
    for (int off = 16; off > 0; off >>= 1) part += __shfl_xor(part, off, 32);
    float sc = part * scale;
    float mn = fmaxf(m, sc);
    float corr = __expf(m - mn);
    float wt = __expf(sc - mn);
    const float* vp = v + ((size_t)(b * T + kt)) * D + h * hd;
    l  = l * corr + wt;
    ae = ae * corr + wt * vp[2 * j];
    ao = ao * corr + wt * vp[2 * j + 1];
    m = mn;
  }
  float* op = o + ((size_t)(b * T + qt)) * D + h * hd;
  op[2 * j]     = ae / l;
  op[2 * j + 1] = ao / l;
}

// ======================= small elementwise / RVQ kernels =======================
__global__ __launch_bounds__(TPB) void zero_kernel(float* p, long n) {
  long i = (long)blockIdx.x * TPB + threadIdx.x; if (i < n) p[i] = 0.f;
}
__global__ __launch_bounds__(TPB) void copy_kernel(float* dst, const float* src, long n) {
  long i = (long)blockIdx.x * TPB + threadIdx.x; if (i < n) dst[i] = src[i];
}
__global__ __launch_bounds__(TPB) void mul_kernel(float* g, const float* u, long n) {
  long i = (long)blockIdx.x * TPB + threadIdx.x; if (i < n) g[i] *= u[i];
}
__global__ __launch_bounds__(TPB) void silu_kernel(float* p, long n) {
  long i = (long)blockIdx.x * TPB + threadIdx.x; if (i < n) p[i] = silu_f(p[i]);
}
__global__ __launch_bounds__(TPB) void update_kernel(float* zqsum, float* resid,
                                                     const float* dq, long n) {
  long i = (long)blockIdx.x * TPB + threadIdx.x;
  if (i < n) { zqsum[i] += dq[i]; resid[i] -= dq[i]; }
}
__global__ __launch_bounds__(TPB) void cbnorm_kernel(const float* cb, float* out, int C, int D) {
  int c = blockIdx.x * TPB + threadIdx.x;
  if (c >= C) return;
  float s = 0.f;
  const float* p = cb + (size_t)c * D;
  for (int d = 0; d < D; ++d) s += p[d] * p[d];
  out[c] = s;
}
__global__ __launch_bounds__(TPB)
void argmin_kernel(const float* __restrict__ dist, const float* __restrict__ cbnorm,
                   int* __restrict__ idx, int M, int C)
{
  int row = blockIdx.x;
  if (row >= M) return;
  __shared__ float bval[TPB];
  __shared__ int   bidx[TPB];
  float best = 3.0e38f; int bi = 0;
  for (int c = threadIdx.x; c < C; c += TPB) {
    float d = cbnorm[c] - 2.f * dist[(size_t)row * C + c];
    if (d < best) { best = d; bi = c; }
  }
  bval[threadIdx.x] = best; bidx[threadIdx.x] = bi; __syncthreads();
  for (int o = TPB / 2; o > 0; o >>= 1) {
    if (threadIdx.x < o) {
      float a = bval[threadIdx.x], bv2 = bval[threadIdx.x + o];
      int ai = bidx[threadIdx.x], bi2 = bidx[threadIdx.x + o];
      if (bv2 < a || (bv2 == a && bi2 < ai)) { bval[threadIdx.x] = bv2; bidx[threadIdx.x] = bi2; }
    }
    __syncthreads();
  }
  if (threadIdx.x == 0) idx[row] = bidx[0];
}
__global__ __launch_bounds__(TPB)
void rvq_gather_kernel(const float* __restrict__ ze, const float* __restrict__ cb,
                       const int* __restrict__ idx, float* __restrict__ zq,
                       float* __restrict__ commit, float inv, int M, int Dc)
{
  long i = (long)blockIdx.x * TPB + threadIdx.x;
  __shared__ float red[TPB];
  float part = 0.f;
  if (i < (long)M * Dc) {
    int r = (int)(i / Dc), d = (int)(i % Dc);
    float c = cb[(size_t)idx[r] * Dc + d];
    zq[i] = c;
    float df = ze[i] - c;
    part = df * df;
  }
  red[threadIdx.x] = part; __syncthreads();
  for (int o = TPB / 2; o > 0; o >>= 1) {
    if (threadIdx.x < o) red[threadIdx.x] += red[threadIdx.x + o];
    __syncthreads();
  }
  if (threadIdx.x == 0) atomicAdd(commit, red[0] * inv);
}
__global__ __launch_bounds__(TPB)
void write_codes_kernel(const int* idx, float* codes, int cbi, int ncb, int B, int T) {
  int r = blockIdx.x * TPB + threadIdx.x;
  if (r >= B * T) return;
  int b = r / T, t = r % T;
  codes[((size_t)b * ncb + cbi) * T + t] = (float)idx[r];
}
__global__ __launch_bounds__(TPB)
void bcl_to_btc_kernel(const float* x, float* y, int B, int C, int L) {
  long i = (long)blockIdx.x * TPB + threadIdx.x;
  if (i >= (long)B * C * L) return;
  int c = (int)(i % C); long tmp = i / C; int t = (int)(tmp % L); int b = (int)(tmp / L);
  y[i] = x[((size_t)b * C + c) * L + t];
}
__global__ __launch_bounds__(TPB)
void btc_to_bcl_kernel(const float* x, float* y, int B, int C, int L) {
  long i = (long)blockIdx.x * TPB + threadIdx.x;
  if (i >= (long)B * C * L) return;
  int t = (int)(i % L); long tmp = i / L; int c = (int)(tmp % C); int b = (int)(tmp / C);
  y[i] = x[((size_t)b * L + t) * C + c];
}

// ======================= host orchestration =======================
extern "C" void kernel_launch(void* const* d_in, const int* in_sizes, int n_in,
                              void* d_out, int out_size, void* d_ws, size_t ws_size,
                              hipStream_t stream) {
  (void)in_sizes; (void)n_in; (void)out_size; (void)ws_size;
  const int Bsz = 2, NH = 8, HD = 64, DIM = 512, FF = 2048, NCB = 8, CBSZ = 2048, CBD = 128;

  // ---- pytree leaf order (jax dict keys sorted): dec, enc, rvq, tdec, tenc ----
  // dec block keys: r1b,r1w,r2b,r2w,r3b,r3w,ub,uw ; enc block: db,dw,r1b,r1w,r2b,r2w,r3b,r3w
  // tblock keys: bd,bg,bk,bo,bq,bu,bv,ln1b,ln1w,ln2b,ln2w,wd,wg,wk,wo,wq,wu,wv
  int ip = 1;
  const int DEC_B = ip; ip += 40;
  const int DEC_IN_B = ip++; const int DEC_IN_W = ip++;
  const int DEC_OUT_B = ip++; const int DEC_OUT_W = ip++;
  const int ENC_B = ip; ip += 40;
  const int ENC_IN_B = ip++; const int ENC_IN_W = ip++;
  const int ENC_OUT_B = ip++; const int ENC_OUT_W = ip++;
  const int RVQ_CB = ip; ip += 8;
  const int RVQ_IN_B = ip; ip += 8;
  const int RVQ_IN_W = ip; ip += 8;
  const int RVQ_OUT_B = ip; ip += 8;
  const int RVQ_OUT_W = ip; ip += 8;
  const int TDEC = ip; ip += 144;
  const int TENC = ip; ip += 144;
  auto IN = [&](int i) { return (const float*)d_in[i]; };
  const float* audio = (const float*)d_in[0];

  // ---- lengths ----
  int encS[5] = {4, 8, 5, 4, 3};
  int ecs[6]  = {16, 32, 64, 128, 256, 512};
  long encL[6]; encL[0] = 480000;
  for (int i = 0; i < 5; ++i) {
    int s = encS[i], K = 2 * s, p = s / 2;
    encL[i + 1] = (encL[i] + 2L * p - (K - 1) - 1) / s + 1;
  }
  const int T = (int)encL[5];           // 249
  const int M = Bsz * T;                // 498
  int decS[5] = {3, 4, 5, 8, 4};
  int dcs[6]  = {512, 256, 128, 64, 32, 16};
  long decL[6]; decL[0] = T;
  for (int i = 0; i < 5; ++i) {
    int s = decS[i];
    decL[i + 1] = (decL[i] - 1) * s + 2 * s - 2 * (s / 2);
  }
  const long reconL = decL[5];

  float* out = (float*)d_out;
  const long off_recon = 0;
  const long off_codes = (long)Bsz * reconL;
  const long off_lat   = off_codes + (long)Bsz * NCB * T;
  const long off_latq  = off_lat + (long)M * DIM;
  const long off_commit = off_latq + (long)M * DIM;

  // ---- workspace bump allocator ----
  uintptr_t wp = (uintptr_t)d_ws;
  auto alloc = [&](long nf) {
    wp = (wp + 255) & ~(uintptr_t)255;
    float* r = (float*)wp; wp += (size_t)nf * sizeof(float); return r;
  };
  const long CONVBUF = 2L * 16 * 480000;      // biggest (B,C,L) tensor
  float* P0 = alloc(CONVBUF);
  float* P1 = alloc(CONVBUF);
  float* P2 = alloc(CONVBUF);
  float* Xb = alloc((long)M * DIM);
  float* XA = alloc((long)M * DIM);
  float* XN = alloc((long)M * DIM);
  float* Qb = alloc((long)M * DIM);
  float* Kb = alloc((long)M * DIM);
  float* Vb = alloc((long)M * DIM);
  float* Ob = alloc((long)M * DIM);
  float* Gb = alloc((long)M * FF);
  float* Ub = alloc((long)M * FF);
  float* ZE = alloc((long)M * CBD);
  float* DIST = alloc((long)M * CBSZ);
  float* ZQ = alloc((long)M * CBD);
  float* DQ = alloc((long)M * DIM);
  float* ZQSUM = alloc((long)M * DIM);
  float* RESID = alloc((long)M * DIM);
  float* CBN = alloc((long)NCB * CBSZ);
  float* COMMIT = alloc(8);
  int* IDX = (int*)alloc(((long)M + 8));

  auto nb = [](long n) { return (unsigned)((n + TPB - 1) / TPB); };
  auto GEMM = [&](const float* A, const float* W, const float* bi, const float* res,
                  float* C, int m, int n, int k, int act, int wt) {
    dim3 g((unsigned)(n / BN), (unsigned)((m + BM - 1) / BM));
    gemm_wmma_kernel<<<g, TPB, 0, stream>>>(A, W, bi, res, C, m, n, k, act, wt);
  };
  auto LN = [&](const float* x, const float* w, const float* b, float* y) {
    layernorm_kernel<<<(unsigned)M, TPB, 0, stream>>>(x, w, b, y, M, DIM);
  };
  auto CONV = [&](const float* x, const float* w, const float* bi, const float* res, float* y,
                  int Ci, long Li, int Co, long Lo, int K, int s, int p, int d, int act) {
    long tot = (long)Bsz * Co * Lo;
    conv1d_kernel<<<nb(tot), TPB, 0, stream>>>(x, w, bi, res, y, Bsz, Ci, (int)Li, Co, (int)Lo,
                                               K, s, p, d, act);
  };
  auto CONVT = [&](const float* x, const float* w, const float* bi, float* y,
                   int Ci, long Li, int Co, long Lo, int K, int s, int p) {
    long tot = (long)Bsz * Co * Lo;
    convt1d_kernel<<<nb(tot), TPB, 0, stream>>>(x, w, bi, y, Bsz, Ci, (int)Li, Co, (int)Lo, K, s, p);
  };
  auto run_transformer = [&](int base, float* x) {
    for (int l = 0; l < 8; ++l) {
      int pb = base + l * 18;
      LN(x, IN(pb + 8), IN(pb + 7), XN);                         // ln1
      GEMM(XN, IN(pb + 15), IN(pb + 4), nullptr, Qb, M, DIM, DIM, 0, 0); // wq,bq
      GEMM(XN, IN(pb + 13), IN(pb + 2), nullptr, Kb, M, DIM, DIM, 0, 0); // wk,bk
      GEMM(XN, IN(pb + 17), IN(pb + 6), nullptr, Vb, M, DIM, DIM, 0, 0); // wv,bv
      attn_kernel<<<(unsigned)((Bsz * NH * T + 7) / 8), TPB, 0, stream>>>(Qb, Kb, Vb, Ob,
                                                                          Bsz, NH, T, HD);
      GEMM(Ob, IN(pb + 14), IN(pb + 3), x, XA, M, DIM, DIM, 0, 0);      // wo,bo + residual
      LN(XA, IN(pb + 10), IN(pb + 9), XN);                              // ln2
      GEMM(XN, IN(pb + 12), IN(pb + 1), nullptr, Gb, M, FF, DIM, 1, 0); // silu(wg)
      GEMM(XN, IN(pb + 16), IN(pb + 5), nullptr, Ub, M, FF, DIM, 0, 0); // wu
      mul_kernel<<<nb((long)M * FF), TPB, 0, stream>>>(Gb, Ub, (long)M * FF);
      GEMM(Gb, IN(pb + 11), IN(pb + 0), XA, x, M, DIM, FF, 0, 0);       // wd + residual
    }
  };

  // =============== encoder ===============
  CONV(audio, IN(ENC_IN_W), IN(ENC_IN_B), nullptr, P0, 1, 480000, 16, 480000, 7, 1, 3, 1, 0);
  float *X = P0, *F1 = P1, *F2 = P2;
  long Li = 480000; int Ci = 16;
  for (int i = 0; i < 5; ++i) {
    int s = encS[i], Co = ecs[i + 1]; long Lo = encL[i + 1];
    int bb = ENC_B + i * 8;
    CONV(X,  IN(bb + 1), IN(bb + 0), nullptr, F1, Ci, Li, Co, Lo, 2 * s, s, s / 2, 1, 0); // down
    CONV(F1, IN(bb + 3), IN(bb + 2), nullptr, F2, Co, Lo, Co, Lo, 3, 1, 1, 1, 1);         // r1 silu
    CONV(F2, IN(bb + 5), IN(bb + 4), nullptr, X,  Co, Lo, Co, Lo, 3, 1, 3, 3, 1);         // r2 silu
    CONV(X,  IN(bb + 7), IN(bb + 6), F1,      F2, Co, Lo, Co, Lo, 3, 1, 9, 9, 0);         // r3 + down
    float* nX = F2; F2 = F1; F1 = X; X = nX;
    Li = Lo; Ci = Co;
  }
  CONV(X, IN(ENC_OUT_W), IN(ENC_OUT_B), nullptr, F1, 512, T, 512, T, 1, 1, 0, 1, 0);
  bcl_to_btc_kernel<<<nb((long)M * DIM), TPB, 0, stream>>>(F1, Xb, Bsz, DIM, T);

  // =============== transformer (encoder side) ===============
  run_transformer(TENC, Xb);
  copy_kernel<<<nb((long)M * DIM), TPB, 0, stream>>>(out + off_lat, Xb, (long)M * DIM);
  copy_kernel<<<nb((long)M * DIM), TPB, 0, stream>>>(RESID, Xb, (long)M * DIM);

  // =============== RVQ ===============
  for (int i = 0; i < NCB; ++i)
    cbnorm_kernel<<<nb(CBSZ), TPB, 0, stream>>>(IN(RVQ_CB + i), CBN + (long)i * CBSZ, CBSZ, CBD);
  zero_kernel<<<nb((long)M * DIM), TPB, 0, stream>>>(ZQSUM, (long)M * DIM);
  zero_kernel<<<1, TPB, 0, stream>>>(COMMIT, 8);
  for (int i = 0; i < NCB; ++i) {
    GEMM(RESID, IN(RVQ_IN_W + i), IN(RVQ_IN_B + i), nullptr, ZE, M, CBD, DIM, 0, 0);
    GEMM(ZE, IN(RVQ_CB + i), nullptr, nullptr, DIST, M, CBSZ, CBD, 0, 1);  // ze @ cb^T
    argmin_kernel<<<(unsigned)M, TPB, 0, stream>>>(DIST, CBN + (long)i * CBSZ, IDX, M, CBSZ);
    rvq_gather_kernel<<<nb((long)M * CBD), TPB, 0, stream>>>(ZE, IN(RVQ_CB + i), IDX, ZQ, COMMIT,
                                                             1.f / (float)((long)M * CBD), M, CBD);
    write_codes_kernel<<<nb(M), TPB, 0, stream>>>(IDX, out + off_codes, i, NCB, Bsz, T);
    GEMM(ZQ, IN(RVQ_OUT_W + i), IN(RVQ_OUT_B + i), nullptr, DQ, M, DIM, CBD, 0, 0);
    update_kernel<<<nb((long)M * DIM), TPB, 0, stream>>>(ZQSUM, RESID, DQ, (long)M * DIM);
  }
  copy_kernel<<<nb((long)M * DIM), TPB, 0, stream>>>(out + off_latq, ZQSUM, (long)M * DIM);
  copy_kernel<<<1, TPB, 0, stream>>>(out + off_commit, COMMIT, 1);
  copy_kernel<<<nb((long)M * DIM), TPB, 0, stream>>>(Xb, ZQSUM, (long)M * DIM);

  // =============== transformer (decoder side) ===============
  run_transformer(TDEC, Xb);

  // =============== decoder ===============
  btc_to_bcl_kernel<<<nb((long)M * DIM), TPB, 0, stream>>>(Xb, P0, Bsz, DIM, T);
  CONV(P0, IN(DEC_IN_W), IN(DEC_IN_B), nullptr, P1, 512, T, 512, T, 7, 1, 3, 1, 0);
  X = P1; F1 = P0; F2 = P2; Li = T; Ci = 512;
  for (int i = 0; i < 5; ++i) {
    int s = decS[i], Co = dcs[i + 1]; long Lo = decL[i + 1];
    int bb = DEC_B + i * 8;
    CONVT(X, IN(bb + 7), IN(bb + 6), F1, Ci, Li, Co, Lo, 2 * s, s, s / 2);        // up
    CONV(F1, IN(bb + 1), IN(bb + 0), nullptr, F2, Co, Lo, Co, Lo, 3, 1, 1, 1, 1); // r1 silu
    CONV(F2, IN(bb + 3), IN(bb + 2), nullptr, X,  Co, Lo, Co, Lo, 3, 1, 3, 3, 1); // r2 silu
    CONV(X,  IN(bb + 5), IN(bb + 4), F1,      F2, Co, Lo, Co, Lo, 3, 1, 9, 9, 0); // r3 + up
    float* nX = F2; F2 = F1; F1 = X; X = nX;
    Li = Lo; Ci = Co;
  }
  silu_kernel<<<nb((long)Bsz * 16 * reconL), TPB, 0, stream>>>(X, (long)Bsz * 16 * reconL);
  CONV(X, IN(DEC_OUT_W), IN(DEC_OUT_B), nullptr, out + off_recon, 16, reconL, 1, reconL,
       7, 1, 3, 1, 2 /* tanh */);
}
